// Agent_45535243272417
// MI455X (gfx1250) — matrix operand order repaired and checked by reading, not compiled, not run on hardware
//
#include <hip/hip_runtime.h>
#include <stdint.h>

// ---------------- problem constants ----------------
#define BB     128
#define HWD    84
#define HID    1024
#define MEMN   127
#define LSEQ   128      // MEMN + 1
#define DSTATE 16
#define DCONV  4
#define DIN    2048     // HID * EXPAND
#define DTRANK 64       // HID / 16
#define NACT   18
#define MROWS  (BB * LSEQ)   // 16384
#define XDBLP  128           // padded x_dbl row stride (96 -> 128)

// ---------------- types for WMMA ----------------
typedef __attribute__((ext_vector_type(16))) __bf16       v16bf;
typedef __attribute__((ext_vector_type(8)))  float        v8f;
typedef __attribute__((ext_vector_type(4)))  unsigned int u32x4;

union FragBF { v16bf v; u32x4 q[2]; };

__device__ __forceinline__ unsigned short f2bf(float f) {
    unsigned u = __float_as_uint(f);
    unsigned r = u + 0x7FFFu + ((u >> 16) & 1u);  // round-to-nearest-even
    return (unsigned short)(r >> 16);
}
__device__ __forceinline__ float bf2f(unsigned short h) {
    return __uint_as_float(((unsigned)h) << 16);
}

// Async global->LDS 16B load. The LDS pointer is passed as a (dummy) operand so
// the shared object ESCAPES -> the "memory" clobber forces later ds_loads to be
// real (otherwise LLVM folds reads of never-visibly-written LDS to undef).
__device__ __forceinline__ void async_ld_b128(unsigned short* lds_dst, const unsigned short* g) {
    unsigned off = (unsigned)(uintptr_t)lds_dst;   // low 32 bits = LDS byte offset
    asm volatile("global_load_async_to_lds_b128 %0, %1, off"
                 :: "v"(off), "v"((const void*)g), "v"((void*)lds_dst)
                 : "memory");
}

// ---------------- elementwise helpers ----------------
__global__ void k_f2bf(const float* __restrict__ s, unsigned short* __restrict__ d, unsigned n) {
    unsigned i = blockIdx.x * blockDim.x + threadIdx.x;
    if (i < n) d[i] = f2bf(s[i]);
}
__global__ void k_zero16(unsigned short* __restrict__ p, unsigned n) {
    unsigned i = blockIdx.x * blockDim.x + threadIdx.x;
    if (i < n) p[i] = 0;
}

// ---------------- encoder convs (f32, small) ----------------
__global__ void k_conv1(const float* __restrict__ x, const float* __restrict__ w,
                        const float* __restrict__ b, float* __restrict__ o) {
    unsigned i = blockIdx.x * blockDim.x + threadIdx.x;
    if (i >= (unsigned)BB * 32u * 400u) return;
    unsigned bb = i / (32u * 400u), r = i % (32u * 400u);
    unsigned c = r / 400u, p = r % 400u;
    unsigned oy = p / 20u, ox = p % 20u;
    const float* xb = x + (size_t)bb * 84 * 84;
    const float* wc = w + c * 64;
    float acc = 0.f;
    for (int ky = 0; ky < 8; ++ky) {
        unsigned iy = oy * 4 + ky;
        #pragma unroll
        for (int kx = 0; kx < 8; ++kx)
            acc += wc[ky * 8 + kx] * xb[iy * 84 + ox * 4 + kx];
    }
    acc = acc * (1.f / 255.f) + b[c];
    o[i] = fmaxf(acc, 0.f);
}

__global__ void k_conv2(const float* __restrict__ h1, const float* __restrict__ w,
                        const float* __restrict__ b, float* __restrict__ o) {
    unsigned i = blockIdx.x * blockDim.x + threadIdx.x;
    if (i >= (unsigned)BB * 64u * 81u) return;
    unsigned bb = i / (64u * 81u), r = i % (64u * 81u);
    unsigned c = r / 81u, p = r % 81u;
    unsigned oy = p / 9u, ox = p % 9u;
    const float* hb = h1 + (size_t)bb * 32 * 400;
    float acc = b[c];
    for (int ci = 0; ci < 32; ++ci) {
        const float* wc = w + ((c * 32 + ci) * 16);
        const float* hc = hb + ci * 400;
        #pragma unroll
        for (int ky = 0; ky < 4; ++ky) {
            unsigned iy = oy * 2 + ky;
            #pragma unroll
            for (int kx = 0; kx < 4; ++kx)
                acc += wc[ky * 4 + kx] * hc[iy * 20 + ox * 2 + kx];
        }
    }
    o[i] = fmaxf(acc, 0.f);
}

__global__ void k_conv3(const float* __restrict__ h2, const float* __restrict__ w,
                        const float* __restrict__ b, float* __restrict__ o) {
    unsigned i = blockIdx.x * blockDim.x + threadIdx.x;
    if (i >= (unsigned)BB * 64u * 49u) return;
    unsigned bb = i / (64u * 49u), r = i % (64u * 49u);
    unsigned c = r / 49u, p = r % 49u;
    unsigned oy = p / 7u, ox = p % 7u;
    const float* hb = h2 + (size_t)bb * 64 * 81;
    float acc = b[c];
    for (int ci = 0; ci < 64; ++ci) {
        const float* wc = w + ((c * 64 + ci) * 9);
        const float* hc = hb + ci * 81;
        #pragma unroll
        for (int ky = 0; ky < 3; ++ky) {
            #pragma unroll
            for (int kx = 0; kx < 3; ++kx)
                acc += wc[ky * 3 + kx] * hc[(oy + ky) * 9 + (ox + kx)];
        }
    }
    o[i] = fmaxf(acc, 0.f);
}

// ---------------- bf16 WMMA GEMM: C[M,N] = A[M,K] * W[N,K]^T (+bias, +relu) ----
// Requirements (guaranteed by launcher): M % 128 == 0, N % 64 == 0, K % 64 == 0.
// Block: 256 threads = 8 waves; each wave owns 16 rows x 64 cols (4 accumulators).
// Weight chunk (64 N-rows x 64 K) staged to LDS via async global->LDS loads,
// double-buffered, pipelined with s_wait_asynccnt. A fragments direct from global.
__global__ void k_gemm_bf16(const unsigned short* __restrict__ A,
                            const unsigned short* __restrict__ W,
                            float* __restrict__ C,
                            const float* __restrict__ bias,
                            int M, int N, int K, int act) {
    __shared__ alignas(16) unsigned short smem[2 * 64 * 64];   // 2 x 8KB bf16 tile
    const unsigned short* sm = smem;

    const int lane = threadIdx.x & 31;
    const int wave = threadIdx.x >> 5;
    const int m0 = blockIdx.x * 128 + wave * 16;
    const int nbase = blockIdx.y * 64;
    const int half = lane >> 4;
    const int l15  = lane & 15;

    const unsigned short* arow = A + (size_t)(m0 + l15) * K;

    // cooperative staging: 256 threads x 2 x 16B = 8KB = 64 rows x 128B
    const unsigned rr = threadIdx.x >> 3;   // 0..31 (row within chunk)
    const unsigned seg = threadIdx.x & 7;   // 0..7  (16B segment within row)
    const unsigned short* wbase = W + (size_t)(nbase + rr) * K + seg * 8;
    const size_t wstep32 = (size_t)32 * K;

    v8f acc[4] = {};
    const int nchunks = K >> 6;

    auto stage = [&](int sbuf, int kc) {
        unsigned idx = (unsigned)sbuf * 4096u + rr * 64u + seg * 8u;
        async_ld_b128(&smem[idx], wbase + kc);
        async_ld_b128(&smem[idx + 32u * 64u], wbase + kc + wstep32);
    };
    auto compute = [&](int c) {
        const int kc = c << 6;
        const unsigned sb = (unsigned)(c & 1) * 4096u;
        #pragma unroll
        for (int kk = 0; kk < 64; kk += 32) {
            // A fragment (global, 2 x 16B)
            FragBF a;
            a.q[0] = *(const u32x4*)(arow + kc + kk + 8 * half);        // K = +8h .. +7
            a.q[1] = *(const u32x4*)(arow + kc + kk + 16 + 8 * half);   // K = +16+8h ..
            // all four B fragments first (independent registers -> ds_load clause)
            FragBF b0, b1, b2, b3;
            {
                unsigned base = sb + (unsigned)(l15) * 64u + (unsigned)(kk + 8 * half);
                b0.q[0] = *(const u32x4*)(sm + base);
                b0.q[1] = *(const u32x4*)(sm + base + 16);
                b1.q[0] = *(const u32x4*)(sm + base + 16u * 64u);
                b1.q[1] = *(const u32x4*)(sm + base + 16u * 64u + 16);
                b2.q[0] = *(const u32x4*)(sm + base + 32u * 64u);
                b2.q[1] = *(const u32x4*)(sm + base + 32u * 64u + 16);
                b3.q[0] = *(const u32x4*)(sm + base + 48u * 64u);
                b3.q[1] = *(const u32x4*)(sm + base + 48u * 64u + 16);
            }
            // then four back-to-back WMMAs on independent accumulators
            acc[0] = __builtin_amdgcn_wmma_f32_16x16x32_bf16(
                         false, a.v, false, b0.v, (short)0, acc[0], false, false);
            acc[1] = __builtin_amdgcn_wmma_f32_16x16x32_bf16(
                         false, a.v, false, b1.v, (short)0, acc[1], false, false);
            acc[2] = __builtin_amdgcn_wmma_f32_16x16x32_bf16(
                         false, a.v, false, b2.v, (short)0, acc[2], false, false);
            acc[3] = __builtin_amdgcn_wmma_f32_16x16x32_bf16(
                         false, a.v, false, b3.v, (short)0, acc[3], false, false);
        }
    };

    stage(0, 0);                                   // preload chunk 0
    for (int c = 0; c < nchunks - 1; ++c) {        // branch-free steady state
        stage((c + 1) & 1, (c + 1) << 6);
        __builtin_prefetch((const void*)(arow + ((c + 1) << 6)), 0, 1);
        asm volatile("s_wait_asynccnt 0x2" ::: "memory");  // chunk c done, c+1 in flight
        __syncthreads();
        compute(c);
        __syncthreads();
    }
    asm volatile("s_wait_asynccnt 0x0" ::: "memory");
    __syncthreads();
    compute(nchunks - 1);

    #pragma unroll
    for (int t = 0; t < 4; ++t) {
        int nc = nbase + t * 16 + l15;
        float bv = bias ? bias[nc] : 0.f;
        #pragma unroll
        for (int r = 0; r < 8; ++r) {
            float v = acc[t][r] + bv;
            if (act) v = fmaxf(v, 0.f);
            C[(size_t)(m0 + r + 8 * half) * N + nc] = v;
        }
    }
}

// ---------------- sequence build: seq[b,l,:] = (l<127? prev : cur), bf16 -------
__global__ void k_build_seq(const float* __restrict__ prev, const float* __restrict__ cur,
                            unsigned short* __restrict__ seq) {
    unsigned i = blockIdx.x * blockDim.x + threadIdx.x;
    if (i >= (unsigned)MROWS * HID) return;
    unsigned row = i >> 10, k = i & 1023u;
    unsigned b = row >> 7, l = row & 127u;
    float v = (l < 127u) ? prev[(size_t)(b * 127u + l) * HID + k]
                         : cur[(size_t)b * HID + k];
    seq[i] = f2bf(v);
}

// ---------------- depthwise causal conv1d (pad 3) + SiLU -> xs bf16 -----------
__global__ void k_conv1d_silu(const float* __restrict__ xin, const float* __restrict__ w,
                              const float* __restrict__ bias, unsigned short* __restrict__ xs) {
    unsigned i = blockIdx.x * blockDim.x + threadIdx.x;
    if (i >= (unsigned)MROWS * DIN) return;
    unsigned row = i >> 11;           // b*128 + l
    unsigned d = i & 2047u;
    unsigned b = row >> 7, l = row & 127u;
    float acc = bias[d];
    #pragma unroll
    for (int t = 0; t < DCONV; ++t) {
        int ll = (int)l - (DCONV - 1) + t;
        if (ll >= 0)
            acc += w[d * DCONV + t] * xin[((size_t)(b * LSEQ + ll) << 11) + d];
    }
    float s = acc / (1.f + __expf(-acc));   // silu
    xs[i] = f2bf(s);
}

__global__ void k_slice_dtr(const float* __restrict__ xdbl, unsigned short* __restrict__ dtr) {
    unsigned i = blockIdx.x * blockDim.x + threadIdx.x;
    if (i >= (unsigned)MROWS * DTRANK) return;
    unsigned row = i / DTRANK, c = i % DTRANK;
    dtr[i] = f2bf(xdbl[(size_t)row * XDBLP + c]);
}

__global__ void k_softplus(float* __restrict__ dt, unsigned n) {
    unsigned i = blockIdx.x * blockDim.x + threadIdx.x;
    if (i >= n) return;
    float v = dt[i];
    dt[i] = (v > 20.f) ? v : log1pf(__expf(v));
}

__global__ void k_aneg(const float* __restrict__ Alog, float* __restrict__ Aneg, unsigned n) {
    unsigned i = blockIdx.x * blockDim.x + threadIdx.x;
    if (i < n) Aneg[i] = -__expf(Alog[i]);
}

// ---------------- selective scan: thread per (b,d), 16-state in registers -----
__global__ void k_scan(const float* __restrict__ dt, const unsigned short* __restrict__ xs,
                       const float* __restrict__ xdbl, const float* __restrict__ Aneg,
                       const float* __restrict__ Dvec, const float* __restrict__ zlast,
                       unsigned short* __restrict__ ygate) {
    const int b = blockIdx.y;
    const int d = blockIdx.x * blockDim.x + threadIdx.x;
    __shared__ float sB[DSTATE], sC[DSTATE];
    float h[DSTATE], Av[DSTATE];
    #pragma unroll
    for (int s = 0; s < DSTATE; ++s) { h[s] = 0.f; Av[s] = Aneg[d * DSTATE + s]; }
    float y = 0.f, xlast = 0.f;
    for (int l = 0; l < LSEQ; ++l) {
        const size_t row = (size_t)(b * LSEQ + l);
        if (threadIdx.x < 2 * DSTATE) {
            int s = threadIdx.x & (DSTATE - 1);
            int col = DTRANK + ((threadIdx.x < DSTATE) ? s : (DSTATE + s));
            float v = xdbl[row * XDBLP + col];
            if (threadIdx.x < DSTATE) sB[s] = v; else sC[s] = v;
        }
        __syncthreads();
        float dtv = dt[row * DIN + d];
        float xv  = bf2f(xs[row * DIN + d]);
        float dtx = dtv * xv;
        #pragma unroll
        for (int s = 0; s < DSTATE; ++s)
            h[s] = h[s] * __expf(dtv * Av[s]) + dtx * sB[s];
        if (l == LSEQ - 1) {
            float acc = 0.f;
            #pragma unroll
            for (int s = 0; s < DSTATE; ++s) acc += h[s] * sC[s];
            y = acc; xlast = xv;
        }
        __syncthreads();
    }
    float yt = y + xlast * Dvec[d];
    float z  = zlast[(size_t)b * DIN + d];
    float sz = z / (1.f + __expf(-z));
    ygate[(size_t)b * DIN + d] = f2bf(yt * sz);
}

// ---------------- heads: logits/value/log_softmax/entropy per batch row -------
__global__ void k_heads(const float* __restrict__ ctx, const float* __restrict__ aw,
                        const float* __restrict__ ab, const float* __restrict__ crw,
                        const float* __restrict__ crb, const int* __restrict__ action,
                        float* __restrict__ out) {
    const int b = blockIdx.x, tid = threadIdx.x;
    __shared__ float red[256];
    float acc[NACT + 1];
    #pragma unroll
    for (int r = 0; r <= NACT; ++r) acc[r] = 0.f;
    for (int k = tid; k < HID; k += 256) {
        float cv = ctx[(size_t)b * HID + k];
        #pragma unroll
        for (int r = 0; r < NACT; ++r) acc[r] += cv * aw[r * HID + k];
        acc[NACT] += cv * crw[k];
    }
    float rowsum[NACT + 1];
    for (int r = 0; r <= NACT; ++r) {
        red[tid] = acc[r];
        __syncthreads();
        for (int s = 128; s > 0; s >>= 1) {
            if (tid < s) red[tid] += red[tid + s];
            __syncthreads();
        }
        if (tid == 0) rowsum[r] = red[0];
        __syncthreads();
    }
    if (tid == 0) {
        float logits[NACT], mx = -1e30f;
        #pragma unroll
        for (int r = 0; r < NACT; ++r) { logits[r] = rowsum[r] + ab[r]; mx = fmaxf(mx, logits[r]); }
        float se = 0.f;
        #pragma unroll
        for (int r = 0; r < NACT; ++r) se += __expf(logits[r] - mx);
        float lse = mx + logf(se);
        float ent = 0.f;
        #pragma unroll
        for (int r = 0; r < NACT; ++r) { float lp = logits[r] - lse; ent -= __expf(lp) * lp; }
        int a = action[b];
        out[b]            = logits[a] - lse;        // log_prob
        out[BB + b]       = ent;                    // entropy
        out[2 * BB + b]   = rowsum[NACT] + crb[0];  // value
    }
}

// ---------------- new_memory = [prev[:,1:,:], context] ------------------------
__global__ void k_newmem(const float* __restrict__ prev, const float* __restrict__ ctx,
                         float* __restrict__ outp) {
    unsigned i = blockIdx.x * blockDim.x + threadIdx.x;
    if (i >= (unsigned)BB * MEMN * HID) return;
    unsigned b = i / (MEMN * HID);
    unsigned r = i % (MEMN * HID);
    unsigned l = r / HID, k = r % HID;
    float v = (l < MEMN - 1) ? prev[(size_t)b * MEMN * HID + (size_t)(l + 1) * HID + k]
                             : ctx[(size_t)b * HID + k];
    outp[i] = v;
}

// ---------------- launcher ----------------------------------------------------
extern "C" void kernel_launch(void* const* d_in, const int* in_sizes, int n_in,
                              void* d_out, int out_size, void* d_ws, size_t ws_size,
                              hipStream_t stream) {
    const float* x       = (const float*)d_in[0];
    const float* prevm   = (const float*)d_in[1];
    const int*   action  = (const int*)d_in[2];
    const float* c1w = (const float*)d_in[3];
    const float* c1b = (const float*)d_in[4];
    const float* c2w = (const float*)d_in[5];
    const float* c2b = (const float*)d_in[6];
    const float* c3w = (const float*)d_in[7];
    const float* c3b = (const float*)d_in[8];
    const float* fcw = (const float*)d_in[9];
    const float* fcb = (const float*)d_in[10];
    const float* inpw = (const float*)d_in[11];
    const float* cw1d = (const float*)d_in[12];
    const float* cb1d = (const float*)d_in[13];
    const float* xpw  = (const float*)d_in[14];
    const float* dtpw = (const float*)d_in[15];
    const float* dtpb = (const float*)d_in[16];
    const float* Alog = (const float*)d_in[17];
    const float* Dv   = (const float*)d_in[18];
    const float* opw  = (const float*)d_in[19];
    const float* aw   = (const float*)d_in[20];
    const float* ab   = (const float*)d_in[21];
    const float* crw  = (const float*)d_in[22];
    const float* crb  = (const float*)d_in[23];
    float* out = (float*)d_out;

    char* ws = (char*)d_ws;
    size_t off = 0;
    auto carve = [&](size_t bytes) -> char* {
        char* p = ws + off;
        off = (off + bytes + 255) & ~(size_t)255;
        return p;
    };
    unsigned short* wInProj = (unsigned short*)carve((size_t)2 * DIN * HID * 2);   // 4096x1024
    unsigned short* wOutProj= (unsigned short*)carve((size_t)HID * DIN * 2);       // 1024x2048
    unsigned short* wXProj  = (unsigned short*)carve((size_t)XDBLP * DIN * 2);     // padded 128x2048
    unsigned short* wDtProj = (unsigned short*)carve((size_t)DIN * DTRANK * 2);
    unsigned short* wFc     = (unsigned short*)carve((size_t)HID * 3136 * 2);
    float* h1   = (float*)carve((size_t)BB * 32 * 400 * 4);
    float* h2   = (float*)carve((size_t)BB * 64 * 81 * 4);
    float* h3   = (float*)carve((size_t)BB * 3136 * 4);
    unsigned short* h3bf = (unsigned short*)carve((size_t)BB * 3136 * 2);
    float* cur  = (float*)carve((size_t)BB * HID * 4);
    unsigned short* curbf = (unsigned short*)carve((size_t)BB * HID * 2);
    unsigned short* seqbf = (unsigned short*)carve((size_t)MROWS * HID * 2);
    float* xin  = (float*)carve((size_t)MROWS * DIN * 4);
    unsigned short* xsbf = (unsigned short*)carve((size_t)MROWS * DIN * 2);
    float* xdbl = (float*)carve((size_t)MROWS * XDBLP * 4);                        // padded stride
    unsigned short* dtrbf = (unsigned short*)carve((size_t)MROWS * DTRANK * 2);
    float* dt   = (float*)carve((size_t)MROWS * DIN * 4);
    float* An   = (float*)carve((size_t)DIN * DSTATE * 4);
    float* zlast= (float*)carve((size_t)BB * DIN * 4);
    unsigned short* ygbf = (unsigned short*)carve((size_t)BB * DIN * 2);
    float* ctx  = (float*)carve((size_t)BB * HID * 4);

    auto gemm = [&](const unsigned short* A, const unsigned short* W, float* C,
                    const float* bias, int M, int N, int K, int act) {
        dim3 g(M / 128, N / 64);
        k_gemm_bf16<<<g, 256, 0, stream>>>(A, W, C, bias, M, N, K, act);
    };
    auto ew = [&](unsigned n) { return dim3((n + 255u) / 256u); };

    // weight conversions (f32 -> bf16); x_proj padded 96 -> 128 rows with zeros
    k_f2bf<<<ew(2u * DIN * HID), 256, 0, stream>>>(inpw, wInProj, 2u * DIN * HID);
    k_f2bf<<<ew((unsigned)HID * DIN), 256, 0, stream>>>(opw, wOutProj, (unsigned)HID * DIN);
    k_f2bf<<<ew(96u * DIN), 256, 0, stream>>>(xpw, wXProj, 96u * DIN);
    k_zero16<<<ew(32u * DIN), 256, 0, stream>>>(wXProj + (size_t)96 * DIN, 32u * DIN);
    k_f2bf<<<ew((unsigned)DIN * DTRANK), 256, 0, stream>>>(dtpw, wDtProj, (unsigned)DIN * DTRANK);
    k_f2bf<<<ew((unsigned)HID * 3136u), 256, 0, stream>>>(fcw, wFc, (unsigned)HID * 3136u);

    // encoder
    k_conv1<<<ew(BB * 32u * 400u), 256, 0, stream>>>(x, c1w, c1b, h1);
    k_conv2<<<ew(BB * 64u * 81u), 256, 0, stream>>>(h1, c2w, c2b, h2);
    k_conv3<<<ew(BB * 64u * 49u), 256, 0, stream>>>(h2, c3w, c3b, h3);
    k_f2bf<<<ew(BB * 3136u), 256, 0, stream>>>(h3, h3bf, BB * 3136u);
    gemm(h3bf, wFc, cur, fcb, BB, HID, 3136, 1);              // fc + relu
    k_f2bf<<<ew(BB * HID), 256, 0, stream>>>(cur, curbf, BB * HID);

    // sequence + in_proj (x half over all timesteps; z half only for last step)
    k_build_seq<<<ew((unsigned)MROWS * HID), 256, 0, stream>>>(prevm, cur, seqbf);
    gemm(seqbf, wInProj, xin, nullptr, MROWS, DIN, HID, 0);                        // xin
    gemm(curbf, wInProj + (size_t)DIN * HID, zlast, nullptr, BB, DIN, HID, 0);     // z_last

    // conv1d + silu, x_proj (padded N=128), dt_proj + softplus
    k_conv1d_silu<<<ew((unsigned)MROWS * DIN), 256, 0, stream>>>(xin, cw1d, cb1d, xsbf);
    gemm(xsbf, wXProj, xdbl, nullptr, MROWS, XDBLP, DIN, 0);
    k_slice_dtr<<<ew((unsigned)MROWS * DTRANK), 256, 0, stream>>>(xdbl, dtrbf);
    gemm(dtrbf, wDtProj, dt, dtpb, MROWS, DIN, DTRANK, 0);
    k_softplus<<<ew((unsigned)MROWS * DIN), 256, 0, stream>>>(dt, (unsigned)MROWS * DIN);

    // selective scan (only final-step y is needed) + gating
    k_aneg<<<ew((unsigned)DIN * DSTATE), 256, 0, stream>>>(Alog, An, (unsigned)DIN * DSTATE);
    k_scan<<<dim3(DIN / 256, BB), 256, 0, stream>>>(dt, xsbf, xdbl, An, Dv, zlast, ygbf);

    // out_proj for the last step only -> context
    gemm(ygbf, wOutProj, ctx, nullptr, BB, HID, DIN, 0);

    // heads + new memory
    k_heads<<<BB, 256, 0, stream>>>(ctx, aw, ab, crw, crb, action, out);
    k_newmem<<<ew((unsigned)BB * MEMN * HID), 256, 0, stream>>>(prevm, ctx, out + 3 * BB);
}
